// DisorderedCausalSelfAttention_48369921688056
// MI455X (gfx1250) — matrix operand order repaired and checked by reading, not compiled, not run on hardware
//
#include <hip/hip_runtime.h>
#include <hip/hip_bf16.h>

// ---------------------------------------------------------------------------
// CDNA5 (gfx1250, wave32) fused causal self-attention with RoPE.
// Matrix math: v_wmma_f32_16x16x32_bf16 (A/B bf16, C/D f32).
// K-tile staging in flash attention: TDM tensor_load_to_lds (async DMA).
// ---------------------------------------------------------------------------

typedef __attribute__((ext_vector_type(16))) __bf16       bf16x16;
typedef __attribute__((ext_vector_type(8)))  float        f32x8;
typedef __attribute__((ext_vector_type(4)))  unsigned int u32x4;
typedef __attribute__((ext_vector_type(8)))  unsigned int u32x8;

#define WMMA_BF16(a, b, c) \
    __builtin_amdgcn_wmma_f32_16x16x32_bf16(false, (a), false, (b), (short)0, (c), false, false)

static constexpr int Bc  = 2;
static constexpr int Tc  = 2048;
static constexpr int Cc  = 1024;
static constexpr int NHc = 16;
static constexpr int Dc  = 64;
static constexpr int RDc = 32;

// ---------------------------------------------------------------------------
// GEMM: C[M,N] = A[M,K] (f32) * B[K,N] (f32) + bias[N] in bf16 WMMA, f32 acc.
// Block tile 128x128, BK=32, 256 threads = 8 waves.  Wave w computes rows
// [16w,16w+16) x 128 cols = 8 WMMA accumulators (8 wmma per wave per K-step).
// ---------------------------------------------------------------------------
__global__ __launch_bounds__(256)
void gemm_bf16_wmma(const float* __restrict__ A, const float* __restrict__ B,
                    const float* __restrict__ bias, float* __restrict__ C,
                    int M, int N, int K) {
    __shared__ __bf16 As[128 * 32];   // row-major [m][k]
    __shared__ __bf16 Bs[128 * 32];   // transposed: [n][k] for contiguous frag reads

    const int tid = threadIdx.x;
    const int lid = tid & 31;
    const int w   = tid >> 5;                  // 0..7
    const int m0  = blockIdx.y * 128;
    const int n0  = blockIdx.x * 128;
    const int mrow = w * 16;

    f32x8 acc[8] = {};

    for (int k0 = 0; k0 < K; k0 += 32) {
        // Stage A tile (128x32) f32 -> bf16
        for (int i = tid; i < 128 * 32; i += 256) {
            int r = i >> 5, c = i & 31;
            As[i] = (__bf16)A[(size_t)(m0 + r) * K + (k0 + c)];
        }
        // Stage B tile (32x128) transposed -> [n][k]
        for (int i = tid; i < 32 * 128; i += 256) {
            int r = i >> 7, c = i & 127;      // r = k-local, c = n-local
            Bs[c * 32 + r] = (__bf16)B[(size_t)(k0 + r) * N + (n0 + c)];
        }
        // Prefetch next K-tile into cache (global_prefetch_b8)
        if (k0 + 32 < K) {
            __builtin_prefetch(&A[(size_t)(m0 + (tid >> 1)) * K + (k0 + 32) + ((tid & 1) << 4)], 0, 1);
            __builtin_prefetch(&B[(size_t)(k0 + 32 + (tid >> 3)) * N + n0 + ((tid & 7) << 4)], 0, 1);
        }
        __syncthreads();

        // A fragment: 16x32 bf16 (ISA lane layout: lanes 0-15 K 0..7/16..23,
        // lanes 16-31 K 8..15/24..31, row m = lane&15).
        bf16x16 afrag;
        {
            int m  = mrow + (lid & 15);
            int kb = (lid >> 4) * 8;
            #pragma unroll
            for (int hh = 0; hh < 16; ++hh) {
                int kk = kb + (hh & 7) + ((hh >> 3) << 4);
                afrag[hh] = As[m * 32 + kk];
            }
        }
        #pragma unroll
        for (int c = 0; c < 8; ++c) {
            // B fragment: 32x16; lane = column n, lanes 0-15 K 0..7/16..23,
            // lanes 16-31 K 8..15/24..31 (same order as A).
            bf16x16 bfrag;
            int n  = c * 16 + (lid & 15);
            int kb = (lid >> 4) * 8;
            #pragma unroll
            for (int hh = 0; hh < 16; ++hh)
                bfrag[hh] = Bs[n * 32 + kb + (hh & 7) + ((hh >> 3) << 4)];
            acc[c] = WMMA_BF16(afrag, bfrag, acc[c]);
        }
        __syncthreads();
    }

    // C layout: VGPR r holds M = r (lanes 0-15) / r+8 (lanes 16-31), N = lane&15
    int mbase = m0 + mrow + ((lid >> 4) << 3);
    int nn    = lid & 15;
    #pragma unroll
    for (int c = 0; c < 8; ++c) {
        int n   = n0 + c * 16 + nn;
        float bv = bias[n];
        #pragma unroll
        for (int r = 0; r < 8; ++r)
            C[(size_t)(mbase + r) * N + n] = acc[c][r] + bv;
    }
}

// ---------------------------------------------------------------------------
// RoPE + per-head bias + head split.  Reads fused qkv (4096 x 3072 f32),
// writes q (with 1/sqrt(D) folded in), k, v as bf16 in (B, NH, T, D) layout.
// ---------------------------------------------------------------------------
__global__ __launch_bounds__(256)
void rope_split(const float* __restrict__ qkv,
                const float* __restrict__ cosb, const float* __restrict__ sinb,
                const float* __restrict__ bQ, const float* __restrict__ bK,
                __bf16* __restrict__ q, __bf16* __restrict__ k,
                __bf16* __restrict__ v) {
    int idx = blockIdx.x * blockDim.x + threadIdx.x;   // over B*T*NH*D = 2^22
    if (idx >= Bc * Tc * NHc * Dc) return;
    int d = idx & (Dc - 1);
    int h = (idx >> 6) & (NHc - 1);
    int t = (idx >> 10) & (Tc - 1);
    int b = idx >> 21;

    const float scale = 0.125f;                        // D^-0.5, folded into q
    const float* row = qkv + (size_t)(b * Tc + t) * (3 * Cc);
    float qv = row[h * Dc + d];
    float kv = row[Cc + h * Dc + d];
    float vv = row[2 * Cc + h * Dc + d];

    if (d < RDc) {
        float c = cosb[(size_t)(b * Tc + t) * RDc + d];
        float s = sinb[(size_t)(b * Tc + t) * RDc + d];
        int   pd  = (d < RDc / 2) ? d + RDc / 2 : d - RDc / 2;
        float sgn = (d < RDc / 2) ? -1.f : 1.f;
        float qp = row[h * Dc + pd];
        float kp = row[Cc + h * Dc + pd];
        qv = qv * c + sgn * qp * s;
        kv = kv * c + sgn * kp * s;
    }
    qv += bQ[h * Dc + d];
    kv += bK[h * Dc + d];

    size_t o = (((size_t)b * NHc + h) * Tc + t) * Dc + d;
    q[o] = (__bf16)(qv * scale);
    k[o] = (__bf16)kv;
    v[o] = (__bf16)vv;
}

// ---------------------------------------------------------------------------
// Flash attention (causal, online softmax).  128 threads = 4 waves; each wave
// owns 16 query rows (block = 64).  Key tiles of 32:
//   - K tile (contiguous 4 KB) staged by the TDM: tensor_load_to_lds,
//     issued by wave 0, waited with s_wait_tensorcnt.
//   - V tile staged transposed by all threads (TDM cannot transpose) -- this
//     overlaps with the TDM DMA.
// S = Q*K^T : 4x v_wmma_f32_16x16x32_bf16 ; O += P*V : 4 more.
// ---------------------------------------------------------------------------
__global__ __launch_bounds__(128)
void flash_attn(const __bf16* __restrict__ Q, const __bf16* __restrict__ K,
                const __bf16* __restrict__ V, float* __restrict__ Y) {
    __shared__ __bf16 Ks[32 * 64];        // [key][d]  row-major (TDM target)
    __shared__ __bf16 Vs[64 * 32];        // [d][key]  transposed
    __shared__ __bf16 Ps[4][16 * 32];     // per-wave P tile, [m][key]

    const int tid = threadIdx.x;
    const int lid = tid & 31;
    const int w   = tid >> 5;             // 0..3
    const int h   = blockIdx.y;
    const int b   = blockIdx.z;
    const int q0blk = blockIdx.x * 64;
    const int q0    = q0blk + w * 16;

    const size_t base = ((size_t)b * NHc + h) * Tc * Dc;
    const __bf16* Qbh = Q + base;
    const __bf16* Kbh = K + base;
    const __bf16* Vbh = V + base;

    const float MASKV = -3.0e38f;

    // Q fragments for D=64 (two K-steps of 32), held in registers.
    bf16x16 qfrag[2];
    {
        int m  = q0 + (lid & 15);
        int kb = (lid >> 4) * 8;
        #pragma unroll
        for (int s = 0; s < 2; ++s)
            #pragma unroll
            for (int hh = 0; hh < 16; ++hh) {
                int d = s * 32 + kb + (hh & 7) + ((hh >> 3) << 4);
                qfrag[s][hh] = Qbh[(size_t)m * Dc + d];
            }
    }

    // TDM D# group 1 (constant across iterations): data_size=2B (code 1),
    // 1D copy of 2048 bf16 elements (= one 32x64 K tile, contiguous).
    u32x8 g1;
    g1[0] = 0x00010000u;            // workgroup_mask=0, data_size=1 (2 bytes)
    g1[1] = (2048u & 0xFFFFu) << 16; // tensor_dim0[15:0] = 2048
    g1[2] = (1u << 16);             // tensor_dim0[31:16]=0, tensor_dim1[15:0]=1
    g1[3] = (2048u << 16);          // tensor_dim1[31:16]=0, tile_dim0 = 2048
    g1[4] = 0u;                     // tile_dim1 = 0 (unused), tile_dim2 = 0
    g1[5] = 2048u;                  // tensor_dim0_stride[31:0]
    g1[6] = 0u;
    g1[7] = 0u;
    const unsigned ldsK = (unsigned)(unsigned long long)(&Ks[0]);

    f32x8 o[4] = {};
    float m_i[8], l_i[8];
    #pragma unroll
    for (int r = 0; r < 8; ++r) { m_i[r] = MASKV; l_i[r] = 0.f; }

    const int ktend = q0blk + 64;
    for (int kt = 0; kt < ktend; kt += 32) {
        __syncthreads();
        if (w == 0) {
            // D# group 0: count=1, lds_addr, 57-bit global addr, type=2
            unsigned long long ga =
                (unsigned long long)(Kbh + (size_t)kt * Dc);
            u32x4 g0;
            g0[0] = 1u;
            g0[1] = ldsK;
            g0[2] = (unsigned)(ga & 0xFFFFFFFFu);
            g0[3] = (unsigned)((ga >> 32) & 0x01FFFFFFu) | (2u << 30);
            asm volatile("tensor_load_to_lds %0, %1"
                         :
                         : "s"(g0), "s"(g1)
                         : "memory");
        }
        // V tile staged transposed by all threads -- overlaps the TDM DMA.
        for (int i = tid; i < 32 * 64; i += 128) {
            int kr = i >> 6, d = i & 63;
            Vs[d * 32 + kr] = Vbh[(size_t)(kt + kr) * Dc + d];
        }
        if (w == 0) __builtin_amdgcn_s_wait_tensorcnt((short)0);
        __syncthreads();

        // S = Q * K^T : 16 x 32 scores (two 16x16 column tiles)
        f32x8 sc[2] = {};
        #pragma unroll
        for (int ct = 0; ct < 2; ++ct) {
            #pragma unroll
            for (int ks = 0; ks < 2; ++ks) {
                bf16x16 bfrag;                 // B = K^T, lane = key, halves = d
                int key = ct * 16 + (lid & 15);
                int db  = ks * 32 + (lid >> 4) * 8;
                #pragma unroll
                for (int hh = 0; hh < 16; ++hh)
                    bfrag[hh] = Ks[key * 64 + db + (hh & 7) + ((hh >> 3) << 4)];
                sc[ct] = WMMA_BF16(qfrag[ks], bfrag, sc[ct]);
            }
        }

        // Online softmax per row (row stats replicated across each 16-lane group)
        #pragma unroll
        for (int r = 0; r < 8; ++r) {
            int qi  = q0 + r + ((lid >> 4) << 3);
            int k0i = kt + (lid & 15);
            float s0 = (k0i > qi)      ? MASKV : sc[0][r];
            float s1 = (k0i + 16 > qi) ? MASKV : sc[1][r];
            float rm = fmaxf(s0, s1);
            rm = fmaxf(rm, __shfl_xor(rm, 1, 32));
            rm = fmaxf(rm, __shfl_xor(rm, 2, 32));
            rm = fmaxf(rm, __shfl_xor(rm, 4, 32));
            rm = fmaxf(rm, __shfl_xor(rm, 8, 32));
            float mnew = fmaxf(m_i[r], rm);
            float corr = __expf(m_i[r] - mnew);
            float p0 = __expf(s0 - mnew);
            float p1 = __expf(s1 - mnew);
            int m = r + ((lid >> 4) << 3);
            Ps[w][m * 32 + (lid & 15)]      = (__bf16)p0;
            Ps[w][m * 32 + 16 + (lid & 15)] = (__bf16)p1;
            float rs = p0 + p1;
            rs += __shfl_xor(rs, 1, 32);
            rs += __shfl_xor(rs, 2, 32);
            rs += __shfl_xor(rs, 4, 32);
            rs += __shfl_xor(rs, 8, 32);
            l_i[r] = l_i[r] * corr + rs;
            m_i[r] = mnew;
            #pragma unroll
            for (int c = 0; c < 4; ++c) o[c][r] *= corr;
        }

        // O += P * V   (wave-private LDS round-trip re-shapes P into A layout)
        bf16x16 pfrag;
        {
            int m  = lid & 15;
            int kb = (lid >> 4) * 8;
            #pragma unroll
            for (int hh = 0; hh < 16; ++hh) {
                int kk = kb + (hh & 7) + ((hh >> 3) << 4);
                pfrag[hh] = Ps[w][m * 32 + kk];
            }
        }
        #pragma unroll
        for (int c = 0; c < 4; ++c) {
            bf16x16 vfrag;                     // lane = d column, halves = key
            int d  = c * 16 + (lid & 15);
            int kb = (lid >> 4) * 8;
            #pragma unroll
            for (int hh = 0; hh < 16; ++hh)
                vfrag[hh] = Vs[d * 32 + kb + (hh & 7) + ((hh >> 3) << 4)];
            o[c] = WMMA_BF16(pfrag, vfrag, o[c]);
        }
    }

    // Normalize and write Y in (B, T, C) layout
    #pragma unroll
    for (int c = 0; c < 4; ++c) {
        int col = h * Dc + c * 16 + (lid & 15);
        #pragma unroll
        for (int r = 0; r < 8; ++r) {
            int t = q0 + r + ((lid >> 4) << 3);
            Y[((size_t)b * Tc + t) * Cc + col] = o[c][r] / l_i[r];
        }
    }
}

// ---------------------------------------------------------------------------
extern "C" void kernel_launch(void* const* d_in, const int* in_sizes, int n_in,
                              void* d_out, int out_size, void* d_ws, size_t ws_size,
                              hipStream_t stream) {
    (void)in_sizes; (void)n_in; (void)out_size; (void)ws_size;

    const float* x        = (const float*)d_in[0];
    const float* rope_cos = (const float*)d_in[1];
    const float* rope_sin = (const float*)d_in[2];
    const float* W_attn   = (const float*)d_in[3];
    const float* b_attn   = (const float*)d_in[4];
    const float* W_proj   = (const float*)d_in[5];
    const float* b_proj   = (const float*)d_in[6];
    const float* bQ       = (const float*)d_in[7];
    const float* bK       = (const float*)d_in[8];
    float* out = (float*)d_out;

    const size_t BT = (size_t)Bc * Tc;                 // 4096
    char* ws = (char*)d_ws;
    float*  qkv = (float*)ws;                          // 4096*3072 f32 = 48 MB
    size_t  off = BT * 3 * Cc * sizeof(float);
    __bf16* qb = (__bf16*)(ws + off); off += BT * Cc * sizeof(__bf16);
    __bf16* kb = (__bf16*)(ws + off); off += BT * Cc * sizeof(__bf16);
    __bf16* vb = (__bf16*)(ws + off); off += BT * Cc * sizeof(__bf16);
    float*  y  = (float*)(ws + off);                   // 4096*1024 f32 = 16 MB

    // 1) qkv = x @ W_attn + b_attn            (4096 x 3072, K=1024)
    gemm_bf16_wmma<<<dim3(3 * Cc / 128, BT / 128), 256, 0, stream>>>(
        x, W_attn, b_attn, qkv, (int)BT, 3 * Cc, Cc);

    // 2) RoPE + biases + head split -> bf16 (B, NH, T, D)
    {
        int total = Bc * Tc * NHc * Dc;
        rope_split<<<(total + 255) / 256, 256, 0, stream>>>(
            qkv, rope_cos, rope_sin, bQ, bK, qb, kb, vb);
    }

    // 3) Causal flash attention -> y (B, T, C) f32
    flash_attn<<<dim3(Tc / 64, NHc, Bc), 128, 0, stream>>>(qb, kb, vb, y);

    // 4) out = y @ W_proj + b_proj            (4096 x 1024, K=1024)
    gemm_bf16_wmma<<<dim3(Cc / 128, BT / 128), 256, 0, stream>>>(
        y, W_proj, b_proj, out, (int)BT, Cc, Cc);
}
